// ShowAttendTellModel_71880572666470
// MI455X (gfx1250) — matrix operand order repaired and checked by reading, not compiled, not run on hardware
//
#include <hip/hip_runtime.h>
#include <math.h>

#define B_ 128
#define T_ 20
#define L_ 196
#define H_ 512
#define E_ 512
#define V_ 10000

typedef unsigned short u16;
typedef __attribute__((ext_vector_type(16))) __bf16          v16bf;
typedef __attribute__((ext_vector_type(8)))  float           v8f;
typedef __attribute__((ext_vector_type(8)))  unsigned short  v8u;
typedef __attribute__((ext_vector_type(16))) unsigned short  v16u;
typedef __attribute__((ext_vector_type(4)))  unsigned int    v4ui;
typedef __attribute__((ext_vector_type(8)))  int             v8i;
typedef __attribute__((ext_vector_type(4)))  int             v4i;

#if __has_builtin(__builtin_amdgcn_tensor_load_to_lds)
#define HAVE_TDM 1
#else
#define HAVE_TDM 0
#endif

__device__ __forceinline__ u16 f2bf(float f) {
  unsigned u = __builtin_bit_cast(unsigned, f);
  u = u + 0x7FFFu + ((u >> 16) & 1u);          // round-to-nearest-even
  return (u16)(u >> 16);
}
__device__ __forceinline__ float bf2f(u16 h) {
  return __builtin_bit_cast(float, ((unsigned)h) << 16);
}
__device__ __forceinline__ float sigm(float x) { return 1.0f / (1.0f + __expf(-x)); }

#if HAVE_TDM
// TDM: DMA a 2D tile (rows x K bf16, row stride K elems) from global into LDS.
// D# layout per CDNA5 ISA ch.8: group0 = {flags, lds_addr, global_addr, type=2},
// group1 = {data_size=2B, tensor dims, tile dims, strides}. 2D -> groups 2/3 zero.
__device__ __forceinline__ void tdm_load_tile(const u16* gptr, unsigned lds_addr,
                                              unsigned rows, unsigned K) {
  unsigned long long ga = (unsigned long long)(uintptr_t)gptr;
  v4ui g0;
  g0[0] = 1u;                                            // count=1, user descriptor
  g0[1] = lds_addr;                                      // LDS byte address
  g0[2] = (unsigned)(ga & 0xFFFFFFFFu);                  // global_addr[31:0]
  g0[3] = (unsigned)((ga >> 32) & 0x01FFFFFFu) | (2u << 30); // addr[56:32] | type=2
  unsigned long long st0 = (unsigned long long)K;        // dim0 stride (elems)
  v8i g1;
  g1[0] = (int)(1u << 16);                               // data_size=1 -> 2 bytes
  g1[1] = (int)((K & 0xFFFFu) << 16);                    // tensor_dim0[15:0] @bit48
  g1[2] = (int)((K >> 16) | ((rows & 0xFFFFu) << 16));   // dim0 hi | tensor_dim1 lo
  g1[3] = (int)((rows >> 16) | (K << 16));               // dim1 hi | tile_dim0=K
  g1[4] = (int)rows;                                     // tile_dim1=rows, tile_dim2=0
  g1[5] = (int)(unsigned)(st0 & 0xFFFFFFFFu);            // tensor_dim0_stride lo
  g1[6] = (int)(unsigned)((st0 >> 32) & 0xFFFFu);        // stride0 hi | stride1 lo=0
  g1[7] = 0;
  v4i z4 = {};
#if defined(__clang_major__) && (__clang_major__ >= 23)
  typedef __attribute__((ext_vector_type(8))) int v8i_t;
  v8i_t z8 = {};
  __builtin_amdgcn_tensor_load_to_lds(g0, g1, z4, z4, z8, 0);
#else
  __builtin_amdgcn_tensor_load_to_lds(g0, g1, z4, z4, 0);
#endif
}
#endif

// ---------------------------------------------------------------- utilities
__global__ void cast_bf16_kernel(const float* __restrict__ src, u16* __restrict__ dst, int n) {
  int i = blockIdx.x * 256 + threadIdx.x;
  if (i < n) dst[i] = f2bf(src[i]);
}

__global__ void transpose_cast_kernel(const float* __restrict__ src, u16* __restrict__ dst,
                                      int rows, int cols) {
  int i = blockIdx.x * 256 + threadIdx.x;
  if (i < rows * cols) {
    int r = i / cols, c = i % cols;
    dst[c * rows + r] = f2bf(src[i]);
  }
}

__global__ void biascat_kernel(const float* bgx, const float* bgh,
                               const float* bih, const float* bhhl, float* out) {
  int j = blockIdx.x * 256 + threadIdx.x;
  if (j < 2560) out[j] = (j < 512) ? (bgx[j] + bgh[j]) : (bih[j - 512] + bhhl[j - 512]);
}

__global__ void zero_kernel(unsigned* p, int n) {
  int i = blockIdx.x * 256 + threadIdx.x;
  if (i < n) p[i] = 0u;
}

__global__ void embed_kernel(const int* __restrict__ cap, const float* __restrict__ table,
                             u16* __restrict__ out) {
  int i = blockIdx.x * 256 + threadIdx.x;
  if (i >= T_ * B_ * E_) return;
  int e = i % E_;
  int b = (i / E_) % B_;
  int t = i / (E_ * B_);
  int tok = cap[b * T_ + t];
  out[i] = f2bf(table[(size_t)tok * E_ + e]);
}

// ---------------------------------------------------------------- WMMA GEMM
// C(MxN,f32 and/or bf16) = A1(MxK) @ W1(NxK)^T [+ A2 @ W2^T] + bias      (K == 512)
// block: 256 thr = 8 waves, tile 32M x 256N; wave tile 16M x 64N (4 accumulators).
// A tile (32 x K bf16 = 32KB) staged into LDS by the Tensor Data Mover once per pass;
// the 4 waves sharing an M-half then read fragments via ds_load_b128 (4x reuse).
// W fragments: one contiguous 32B row chunk per lane (rows are K-contiguous).
__global__ __launch_bounds__(256) void wmma_gemm_kernel(
    const u16* __restrict__ A1, const u16* __restrict__ W1,
    const u16* __restrict__ A2, const u16* __restrict__ W2,
    const float* __restrict__ bias,
    float* __restrict__ Cf, u16* __restrict__ Cbf,
    int N, int K) {
  __shared__ u16 atile[32 * 512];                 // K <= 512
  const int lane  = threadIdx.x & 31;
  const int wave  = threadIdx.x >> 5;
  const int wm    = wave >> 2;                    // 0..1
  const int wn    = wave & 3;                     // 0..3
  const int m0    = blockIdx.y * 32 + wm * 16;
  const int nbase = blockIdx.x * 256 + wn * 64;
  const int nlane = lane & 15;
  const int abase = (lane < 16) ? 0 : 16;
  const int kbase = (lane < 16) ? 0 : 16;

  v8f acc[4] = {};

  const int npass = (A2 != nullptr) ? 2 : 1;
  for (int pass = 0; pass < npass; ++pass) {
    const u16* __restrict__ A = pass ? A2 : A1;
    const u16* __restrict__ W = pass ? W2 : W1;

    __syncthreads();                              // protect atile reuse across passes
#if HAVE_TDM
    if (wave == 0) {
      tdm_load_tile(A + (size_t)(blockIdx.y * 32) * K,
                    (unsigned)(uintptr_t)atile, 32u, (unsigned)K);
      __builtin_amdgcn_s_wait_tensorcnt(0);       // TENSORcnt is per-wave
    }
#else
    // fallback: tile rows are contiguous (full-K rows) -> flat cooperative copy
    {
      const v8u* src = (const v8u*)(A + (size_t)(blockIdx.y * 32) * K);
      v8u* dst = (v8u*)atile;
      for (int i = threadIdx.x; i < (32 * K) / 8; i += 256) dst[i] = src[i];
    }
#endif
    __syncthreads();

    const u16* arow = atile + (wm * 16 + nlane) * K;
    for (int k0 = 0; k0 < K; k0 += 32) {
      __builtin_prefetch(W + (size_t)(nbase + nlane) * K + k0 + 128, 0, 3);
      v8u  alo = *(const v8u*)(arow + k0 + abase);          // ds_load_b128
      v8u  ahi = *(const v8u*)(arow + k0 + abase + 8);
      v16u au  = __builtin_shufflevector(alo, ahi, 0, 1, 2, 3, 4, 5, 6, 7,
                                                   8, 9, 10, 11, 12, 13, 14, 15);
      v16bf af = __builtin_bit_cast(v16bf, au);
#pragma unroll
      for (int j = 0; j < 4; ++j) {
        int n = nbase + j * 16 + nlane;
        if (n >= N) n = N - 1;                    // clamp loads; stores predicated
        v16u bu  = *(const v16u*)(W + (size_t)n * K + k0 + kbase);
        v16bf bf = __builtin_bit_cast(v16bf, bu);
        acc[j] = __builtin_amdgcn_wmma_f32_16x16x32_bf16(
            false, af, false, bf, (short)0, acc[j], false, false);
      }
    }
  }

  // C frag: lane n = lane&15; VGPR r -> m = r + (lane<16 ? 0 : 8)
  const int mrow = m0 + ((lane >> 4) << 3);
#pragma unroll
  for (int j = 0; j < 4; ++j) {
    int n = nbase + j * 16 + nlane;
    if (n < N) {
      float bv = bias ? bias[n] : 0.0f;
#pragma unroll
      for (int r = 0; r < 8; ++r) {
        float v = acc[j][r] + bv;
        size_t idx = (size_t)(mrow + r) * N + n;
        if (Cf)  Cf[idx]  = v;
        if (Cbf) Cbf[idx] = f2bf(v);
      }
    }
  }
}

// ---------------------------------------------------------------- LSTM pointwise
// preact cols: [0,512)=g_pre  [512,1024)=i  [1024,1536)=f  [1536,2048)=g  [2048,2560)=o
__global__ void lstm_pointwise_kernel(const float* __restrict__ preact,
                                      float* __restrict__ h, float* __restrict__ c,
                                      float* __restrict__ s,
                                      u16* __restrict__ h_bf, u16* __restrict__ s_bf) {
  int i = blockIdx.x * 256 + threadIdx.x;
  if (i >= B_ * H_) return;
  int b = i / H_, hh = i % H_;
  const float* row = preact + (size_t)b * 2560;
  float gp = row[hh];
  float ig = row[512 + hh];
  float fg = row[1024 + hh];
  float gg = row[1536 + hh];
  float og = row[2048 + hh];
  float cv = c[i];
  float g  = sigm(gp);
  float sv = g * tanhf(cv);
  float cn = sigm(fg) * cv + sigm(ig) * tanhf(gg);
  float hn = sigm(og) * tanhf(cn);
  c[i] = cn;  h[i] = hn;  s[i] = sv;
  h_bf[i] = f2bf(hn);
  s_bf[i] = f2bf(sv);
}

// ---------------------------------------------------------------- attention
__global__ __launch_bounds__(256) void attn_z_kernel(const u16* __restrict__ ctx_bf,
                                                     const float* __restrict__ hW,
                                                     const float* __restrict__ w_att,
                                                     float* __restrict__ zbuf) {
  int lane = threadIdx.x & 31;
  int row  = blockIdx.x * 8 + (threadIdx.x >> 5);   // b*L + l
  if (row >= B_ * L_) return;
  int b = row / L_;
  const u16*   cp = ctx_bf + (size_t)row * H_ + lane * 16;
  const float* hp = hW + (size_t)b * H_ + lane * 16;
  const float* wp = w_att + lane * 16;
  v8u c0 = *(const v8u*)cp;
  v8u c1 = *(const v8u*)(cp + 8);
  float acc = 0.0f;
#pragma unroll
  for (int k = 0; k < 8; ++k) acc += tanhf(bf2f(c0[k]) + hp[k]) * wp[k];
#pragma unroll
  for (int k = 0; k < 8; ++k) acc += tanhf(bf2f(c1[k]) + hp[8 + k]) * wp[8 + k];
#pragma unroll
  for (int off = 16; off >= 1; off >>= 1) acc += __shfl_xor(acc, off, 32);
  if (lane == 0) zbuf[b * 197 + (row % L_)] = acc;
}

__global__ void attn_outs_kernel(const float* __restrict__ sW, const float* __restrict__ hW,
                                 const float* __restrict__ b_atts,
                                 const float* __restrict__ W_sent,
                                 const float* __restrict__ b_sent,
                                 float* __restrict__ zbuf) {
  int lane = threadIdx.x & 31;
  int b    = blockIdx.x * 8 + (threadIdx.x >> 5);
  if (b >= B_) return;
  float acc = 0.0f;
  for (int hh = lane; hh < H_; hh += 32)
    acc += tanhf(sW[b * H_ + hh] + hW[b * H_ + hh] + b_atts[hh]) * W_sent[hh];
#pragma unroll
  for (int off = 16; off >= 1; off >>= 1) acc += __shfl_xor(acc, off, 32);
  if (lane == 0) zbuf[b * 197 + 196] = acc + b_sent[0];
}

// softmax over 197, weighted feature pooling, blend with s, tanh(+h_new) -> bf16
__global__ __launch_bounds__(256) void softmax_context_kernel(
    const float* __restrict__ zbuf, const float* __restrict__ feats,
    const float* __restrict__ s, const float* __restrict__ h,
    u16* __restrict__ outin_bf) {
  __shared__ float red[256];
  __shared__ float alpha[200];
  int tid = threadIdx.x;
  int b   = blockIdx.x;
  float v = (tid < 197) ? zbuf[b * 197 + tid] : -1e30f;
  red[tid] = v;
  __syncthreads();
  for (int o = 128; o >= 1; o >>= 1) {
    if (tid < o) red[tid] = fmaxf(red[tid], red[tid + o]);
    __syncthreads();
  }
  float mx = red[0];
  __syncthreads();
  float e = (tid < 197) ? __expf(v - mx) : 0.0f;
  red[tid] = e;
  __syncthreads();
  for (int o = 128; o >= 1; o >>= 1) {
    if (tid < o) red[tid] += red[tid + o];
    __syncthreads();
  }
  float inv = 1.0f / red[0];
  if (tid < 197) alpha[tid] = e * inv;
  __syncthreads();
  float beta = alpha[196];
#pragma unroll
  for (int q = 0; q < 2; ++q) {
    int hh = tid + q * 256;
    const float* fp = feats + (size_t)b * L_ * H_ + hh;
    float acc = 0.0f;
    for (int l = 0; l < L_; ++l) acc += fp[(size_t)l * H_] * alpha[l];
    float ctxv = beta * s[b * H_ + hh] + (1.0f - beta) * (acc * (1.0f / (float)L_));
    outin_bf[b * H_ + hh] = f2bf(tanhf(ctxv + h[b * H_ + hh]));
  }
}

// ---------------------------------------------------------------- launcher
static inline int cdiv_i(int a, int b) { return (a + b - 1) / b; }

extern "C" void kernel_launch(void* const* d_in, const int* in_sizes, int n_in,
                              void* d_out, int out_size, void* d_ws, size_t ws_size,
                              hipStream_t stream) {
  (void)in_sizes; (void)n_in; (void)out_size; (void)ws_size;
  const float* features    = (const float*)d_in[0];
  const int*   captions    = (const int*)d_in[1];
  const float* embed_tab   = (const float*)d_in[2];
  const float* image_att_w = (const float*)d_in[3];
  const float* W_hh        = (const float*)d_in[4];
  const float* b_hh        = (const float*)d_in[5];
  const float* weight_att  = (const float*)d_in[6];
  const float* W_gx        = (const float*)d_in[7];
  const float* b_gx        = (const float*)d_in[8];
  const float* W_gh        = (const float*)d_in[9];
  const float* b_gh        = (const float*)d_in[10];
  const float* W_atts      = (const float*)d_in[11];
  const float* b_atts      = (const float*)d_in[12];
  const float* W_sent      = (const float*)d_in[13];
  const float* b_sent      = (const float*)d_in[14];
  const float* W_ih        = (const float*)d_in[15];
  const float* W_hhl       = (const float*)d_in[16];
  const float* b_ih        = (const float*)d_in[17];
  const float* b_hhl       = (const float*)d_in[18];
  const float* W_cls       = (const float*)d_in[19];
  const float* b_cls       = (const float*)d_in[20];
  float* out = (float*)d_out;

  char*  ws  = (char*)d_ws;
  size_t off = 0;
  auto alloc = [&](size_t bytes) -> char* {
    char* p = ws + off;
    off = (off + bytes + 255) & ~(size_t)255;
    return p;
  };

  u16* feat_bf  = (u16*)alloc((size_t)B_ * L_ * H_ * 2);
  u16* ctx_bf   = (u16*)alloc((size_t)B_ * L_ * H_ * 2);
  u16* emb_bf   = (u16*)alloc((size_t)T_ * B_ * E_ * 2);
  u16* wcat1    = (u16*)alloc((size_t)2560 * 512 * 2);   // rows: W_gx | W_ih
  u16* wcat2    = (u16*)alloc((size_t)2560 * 512 * 2);   // rows: W_gh | W_hhl
  u16* whh_bf   = (u16*)alloc((size_t)512 * 512 * 2);
  u16* watts_bf = (u16*)alloc((size_t)512 * 512 * 2);
  u16* iawT_bf  = (u16*)alloc((size_t)512 * 512 * 2);
  u16* wcls_bf  = (u16*)alloc((size_t)V_ * 512 * 2);
  float* biascat = (float*)alloc(2560 * 4);
  float* h   = (float*)alloc((size_t)B_ * H_ * 4);
  float* c   = (float*)alloc((size_t)B_ * H_ * 4);
  float* s   = (float*)alloc((size_t)B_ * H_ * 4);
  u16* h_bf  = (u16*)alloc((size_t)B_ * H_ * 2);
  u16* s_bf  = (u16*)alloc((size_t)B_ * H_ * 2);
  u16* oi_bf = (u16*)alloc((size_t)B_ * H_ * 2);
  float* preact = (float*)alloc((size_t)B_ * 2560 * 4);
  float* hW = (float*)alloc((size_t)B_ * H_ * 4);
  float* sW = (float*)alloc((size_t)B_ * H_ * 4);
  float* zb = (float*)alloc((size_t)B_ * 197 * 4);

  cast_bf16_kernel<<<cdiv_i(B_ * L_ * H_, 256), 256, 0, stream>>>(features, feat_bf, B_ * L_ * H_);
  cast_bf16_kernel<<<cdiv_i(512 * 512, 256), 256, 0, stream>>>(W_gx, wcat1, 512 * 512);
  cast_bf16_kernel<<<cdiv_i(2048 * 512, 256), 256, 0, stream>>>(W_ih, wcat1 + 512 * 512, 2048 * 512);
  cast_bf16_kernel<<<cdiv_i(512 * 512, 256), 256, 0, stream>>>(W_gh, wcat2, 512 * 512);
  cast_bf16_kernel<<<cdiv_i(2048 * 512, 256), 256, 0, stream>>>(W_hhl, wcat2 + 512 * 512, 2048 * 512);
  cast_bf16_kernel<<<cdiv_i(512 * 512, 256), 256, 0, stream>>>(W_hh, whh_bf, 512 * 512);
  cast_bf16_kernel<<<cdiv_i(512 * 512, 256), 256, 0, stream>>>(W_atts, watts_bf, 512 * 512);
  cast_bf16_kernel<<<cdiv_i(V_ * 512, 256), 256, 0, stream>>>(W_cls, wcls_bf, V_ * 512);
  transpose_cast_kernel<<<cdiv_i(512 * 512, 256), 256, 0, stream>>>(image_att_w, iawT_bf, 512, 512);
  biascat_kernel<<<10, 256, 0, stream>>>(b_gx, b_gh, b_ih, b_hhl, biascat);
  embed_kernel<<<cdiv_i(T_ * B_ * E_, 256), 256, 0, stream>>>(captions, embed_tab, emb_bf);
  zero_kernel<<<cdiv_i(B_ * H_, 256), 256, 0, stream>>>((unsigned*)h, B_ * H_);
  zero_kernel<<<cdiv_i(B_ * H_, 256), 256, 0, stream>>>((unsigned*)c, B_ * H_);
  zero_kernel<<<cdiv_i(B_ * H_ / 2, 256), 256, 0, stream>>>((unsigned*)h_bf, B_ * H_ / 2);

  // ctx_enc = features @ image_att_w  (bf16 output; re-read 20x from L2)
  wmma_gemm_kernel<<<dim3(cdiv_i(512, 256), (B_ * L_) / 32), 256, 0, stream>>>(
      feat_bf, iawT_bf, nullptr, nullptr, nullptr, nullptr, ctx_bf, 512, 512);

  for (int t = 0; t < T_; ++t) {
    const u16* xt = emb_bf + (size_t)t * B_ * E_;
    // preact[B,2560] = x@[W_gx;W_ih]^T + h@[W_gh;W_hhl]^T + (b_gx+b_gh | b_ih+b_hhl)
    wmma_gemm_kernel<<<dim3(cdiv_i(2560, 256), B_ / 32), 256, 0, stream>>>(
        xt, wcat1, h_bf, wcat2, biascat, preact, nullptr, 2560, 512);
    lstm_pointwise_kernel<<<cdiv_i(B_ * H_, 256), 256, 0, stream>>>(preact, h, c, s, h_bf, s_bf);
    wmma_gemm_kernel<<<dim3(2, B_ / 32), 256, 0, stream>>>(
        h_bf, whh_bf, nullptr, nullptr, b_hh, hW, nullptr, 512, 512);
    wmma_gemm_kernel<<<dim3(2, B_ / 32), 256, 0, stream>>>(
        s_bf, watts_bf, nullptr, nullptr, nullptr, sW, nullptr, 512, 512);
    attn_z_kernel<<<cdiv_i(B_ * L_, 8), 256, 0, stream>>>(ctx_bf, hW, weight_att, zb);
    attn_outs_kernel<<<cdiv_i(B_, 8), 256, 0, stream>>>(sW, hW, b_atts, W_sent, b_sent, zb);
    softmax_context_kernel<<<B_, 256, 0, stream>>>(zb, features, s, h, oi_bf);
    // logits -> d_out[t*B*V]
    wmma_gemm_kernel<<<dim3(cdiv_i(V_, 256), B_ / 32), 256, 0, stream>>>(
        oi_bf, wcls_bf, nullptr, nullptr, b_cls, out + (size_t)t * B_ * V_, nullptr, V_, 512);
  }
}
